// GAT_18382460027372
// MI455X (gfx1250) — matrix operand order repaired and checked by reading, not compiled, not run on hardware
//
#include <hip/hip_runtime.h>
#include <hip/hip_bf16.h>

// ---------------------------------------------------------------------------
// GAT (8 heads, D=8) + link classifier, MI455X / gfx1250 (wave32, WMMA).
//
// Precision plan: f16 operands + f32 accumulation via v_wmma_f32_16x16x32_f16
// for both GEMMs (x@W and lr@W1). Attention uses exact online softmax over
// adjacency rows staged in LDS (adj read once per row, 8 head-waves share it).
// ---------------------------------------------------------------------------

typedef __attribute__((ext_vector_type(16))) _Float16 v16h;
typedef __attribute__((ext_vector_type(8)))  _Float16 v8h;
typedef __attribute__((ext_vector_type(8)))  float    v8f;

#define GAT_N     3072
#define GAT_FEAT  512
#define GAT_H     8
#define GAT_D     8
#define GAT_HD    64      // H*D
#define GAT_E     100000
#define GAT_ALPHA 0.2f

// ---------------------------------------------------------------------------
// prep: f32 -> f16 copy of x
__global__ void gat_cvt_x(const float* __restrict__ x, _Float16* __restrict__ xh, int n) {
  int i = blockIdx.x * blockDim.x + threadIdx.x;
  if (i < n) xh[i] = (_Float16)x[i];
}

// Pack W[h,f,d] viewed as Wcat[k=f, col=h*8+d] into per-lane WMMA B fragments.
// B (32x16 f16) fragment layout assumption (mirrors ISA sparse-B doc):
//   lane l: col n = l&15 ; element e in 0..15 holds k = 8*(l>>4) + (e&7) + 16*(e>>3).
// Packed as pW[(((ks*4+nt)*32 + lane)*16 + e] so each lane loads one v16h.
__global__ void gat_pack_wcat(const float* __restrict__ W, _Float16* __restrict__ pW) {
  int idx = blockIdx.x * blockDim.x + threadIdx.x;     // 16*4*32*16 = 32768
  if (idx >= 16 * 4 * 32 * 16) return;
  int e    = idx & 15;
  int lane = (idx >> 4) & 31;
  int nt   = (idx >> 9) & 3;
  int ks   = idx >> 11;
  int k    = ks * 32 + 8 * (lane >> 4) + (e & 7) + 16 * (e >> 3);           // 0..511
  int col  = nt * 16 + (lane & 15);                    // 0..63  (= h*8+d)
  int h = col >> 3, d = col & 7;
  pW[idx] = (_Float16)W[((size_t)h * GAT_FEAT + k) * GAT_D + d];
}

// Same fragment packing for W1 [128,64] (4 k-steps).
__global__ void gat_pack_w1(const float* __restrict__ W1, _Float16* __restrict__ pW1) {
  int idx = blockIdx.x * blockDim.x + threadIdx.x;     // 4*4*32*16 = 8192
  if (idx >= 4 * 4 * 32 * 16) return;
  int e    = idx & 15;
  int lane = (idx >> 4) & 31;
  int nt   = (idx >> 9) & 3;
  int ks   = idx >> 11;
  int k    = ks * 32 + 8 * (lane >> 4) + (e & 7) + 16 * (e >> 3);           // 0..127
  int col  = nt * 16 + (lane & 15);                    // 0..63
  pW1[idx] = (_Float16)W1[(size_t)k * 64 + col];
}

// ---------------------------------------------------------------------------
// Wh GEMM: Y[3072,64] = xh[3072,512] @ Wcat[512,64], f16 in / f32 out.
// One wave per 16-row stripe, 4 N-tile accumulators, 16 k-steps of 32.
__global__ void gat_gemm_wh(const _Float16* __restrict__ xh,
                            const _Float16* __restrict__ pW,
                            float* __restrict__ Y) {
  int tid  = threadIdx.x;
  int wave = blockIdx.x * 8 + (tid >> 5);              // 0..191
  int lane = tid & 31;
  int mrow = lane & 15;
  int hf   = lane >> 4;

  v8f c[4] = {};
  const _Float16* arow = xh + (size_t)(wave * 16 + mrow) * GAT_FEAT;

#pragma unroll
  for (int ks = 0; ks < 16; ++ks) {
    int k0 = ks * 32 + 8 * hf;          // A layout: lane half picks K chunks
    v8h lo = *(const v8h*)(arow + k0);
    v8h hi = *(const v8h*)(arow + k0 + 16);
    v16h a;
#pragma unroll
    for (int e = 0; e < 8; ++e) { a[e] = lo[e]; a[e + 8] = hi[e]; }
#pragma unroll
    for (int nt = 0; nt < 4; ++nt) {
      v16h b = *(const v16h*)(pW + ((size_t)(ks * 4 + nt) * 32 + lane) * 16);
      c[nt] = __builtin_amdgcn_wmma_f32_16x16x32_f16(
          false, a, false, b, (short)0, c[nt], false, false);
    }
  }
#pragma unroll
  for (int nt = 0; nt < 4; ++nt)
#pragma unroll
    for (int v = 0; v < 8; ++v)
      Y[(size_t)(wave * 16 + v + 8 * hf) * GAT_HD + nt * 16 + mrow] = c[nt][v];
}

// ---------------------------------------------------------------------------
// Per-(head,node) attention pre-scores: s1 = Wh.a1, s2 = Wh.a2
__global__ void gat_scores(const float* __restrict__ Y,
                           const float* __restrict__ a1,
                           const float* __restrict__ a2,
                           float* __restrict__ s1, float* __restrict__ s2) {
  int t = blockIdx.x * blockDim.x + threadIdx.x;       // H*N = 24576
  if (t >= GAT_H * GAT_N) return;
  int h = t / GAT_N, n = t - h * GAT_N;
  float sa = 0.f, sb = 0.f;
#pragma unroll
  for (int d = 0; d < GAT_D; ++d) {
    float y = Y[(size_t)n * GAT_HD + h * GAT_D + d];
    sa += y * a1[h * GAT_D + d];
    sb += y * a2[h * GAT_D + d];
  }
  s1[t] = sa; s2[t] = sb;
}

// ---------------------------------------------------------------------------
// Attention + aggregation + ELU, one block per node i, one wave per head.
// adj row staged in LDS once; exact online softmax over neighbors.
__global__ void gat_attn(const float* __restrict__ adj,
                         const float* __restrict__ Y,
                         const float* __restrict__ s1,
                         const float* __restrict__ s2,
                         _Float16* __restrict__ embh) {
  __shared__ float adjrow[GAT_N];                       // 12 KB of 320 KB LDS
  int i = blockIdx.x;
  for (int j = threadIdx.x; j < GAT_N; j += 256)
    adjrow[j] = adj[(size_t)i * GAT_N + j];
  __syncthreads();

  int h = threadIdx.x >> 5;
  int lane = threadIdx.x & 31;
  float m = -1e30f, lsum = 0.f, acc[GAT_D] = {};
  float s1i = s1[h * GAT_N + i];

  for (int j = lane; j < GAT_N; j += 32) {
    if (adjrow[j] > 0.f) {
      float z = s1i + s2[h * GAT_N + j];
      z = (z >= 0.f) ? z : GAT_ALPHA * z;               // leaky_relu
      float nm = fmaxf(m, z);
      float sc = __expf(m - nm);
      float p  = __expf(z - nm);
      lsum = lsum * sc + p;
#pragma unroll
      for (int d = 0; d < GAT_D; ++d)
        acc[d] = acc[d] * sc + p * Y[(size_t)j * GAT_HD + h * GAT_D + d];
      m = nm;
    }
  }
  // merge 32 lane-local online-softmax states
#pragma unroll
  for (int mask = 16; mask > 0; mask >>= 1) {
    float mo = __shfl_xor(m, mask);
    float lo = __shfl_xor(lsum, mask);
    float nm = fmaxf(m, mo);
    float sA = __expf(m - nm), sB = __expf(mo - nm);
    lsum = lsum * sA + lo * sB;
#pragma unroll
    for (int d = 0; d < GAT_D; ++d) {
      float ao = __shfl_xor(acc[d], mask);
      acc[d] = acc[d] * sA + ao * sB;
    }
    m = nm;
  }
  if (lane == 0) {
    float inv = (lsum > 0.f) ? 1.f / lsum : 0.f;
#pragma unroll
    for (int d = 0; d < GAT_D; ++d) {
      float v = acc[d] * inv;
      v = (v > 0.f) ? v : (__expf(v) - 1.f);            // ELU
      embh[(size_t)i * GAT_HD + h * GAT_D + d] = (_Float16)v;
    }
  }
}

// ---------------------------------------------------------------------------
// Classifier: per 16-edge tile, h1 = relu([emb[L]||emb[R]] @ W1 + b1) via WMMA
// (K=128 -> 4 k-steps, N=64 -> 4 tiles), then 64x2 logits via butterfly
// reduction, 2-class log-softmax, NLL summed into *acc.
__device__ __forceinline__ v8h gat_lr_chunk(const _Float16* embh, int li, int ri, int k) {
  const _Float16* p = (k < GAT_HD) ? (embh + (size_t)li * GAT_HD + k)
                                   : (embh + (size_t)ri * GAT_HD + (k - GAT_HD));
  return *(const v8h*)p;
}

__global__ void gat_classifier(const _Float16* __restrict__ embh,
                               const _Float16* __restrict__ pW1,
                               const float* __restrict__ b1,
                               const float* __restrict__ W2,
                               const float* __restrict__ b2,
                               const int* __restrict__ L,
                               const int* __restrict__ R,
                               const int* __restrict__ label,
                               float* __restrict__ acc) {
  int wave = blockIdx.x * 8 + (threadIdx.x >> 5);
  if (wave >= GAT_E / 16) return;                       // uniform per wave
  int lane = threadIdx.x & 31;
  int mrow = lane & 15;
  int hf   = lane >> 4;

  __builtin_prefetch(label + wave * 16, 0, 0);          // global_prefetch_b8

  int eA = wave * 16 + mrow;
  int li = L[eA], ri = R[eA];

  v8f c[4] = {};
#pragma unroll
  for (int ks = 0; ks < 4; ++ks) {
    int k0 = ks * 32 + 8 * hf;
    v8h lo = gat_lr_chunk(embh, li, ri, k0);
    v8h hi = gat_lr_chunk(embh, li, ri, k0 + 16);
    v16h a;
#pragma unroll
    for (int e = 0; e < 8; ++e) { a[e] = lo[e]; a[e + 8] = hi[e]; }
#pragma unroll
    for (int nt = 0; nt < 4; ++nt) {
      v16h b = *(const v16h*)(pW1 + ((size_t)(ks * 4 + nt) * 32 + lane) * 16);
      c[nt] = __builtin_amdgcn_wmma_f32_16x16x32_f16(
          false, a, false, b, (short)0, c[nt], false, false);
    }
  }
  // bias + relu, then logits partials (lane owns column n = nt*16 + mrow)
  float p0[8] = {}, p1[8] = {};
#pragma unroll
  for (int nt = 0; nt < 4; ++nt) {
    int n = nt * 16 + mrow;
    float bn = b1[n];
    float w20 = W2[n * 2 + 0], w21 = W2[n * 2 + 1];
#pragma unroll
    for (int v = 0; v < 8; ++v) {
      float t = c[nt][v] + bn;
      t = (t > 0.f) ? t : 0.f;
      p0[v] += t * w20;
      p1[v] += t * w21;
    }
  }
#pragma unroll
  for (int mask = 1; mask <= 8; mask <<= 1)
#pragma unroll
    for (int v = 0; v < 8; ++v) {
      p0[v] += __shfl_xor(p0[v], mask);
      p1[v] += __shfl_xor(p1[v], mask);
    }
  float b20 = b2[0], b21 = b2[1];
  float s = 0.f;
#pragma unroll
  for (int v = 0; v < 8; ++v) {
    int e = wave * 16 + v + 8 * hf;
    float l0 = p0[v] + b20, l1 = p1[v] + b21;
    float mx = fmaxf(l0, l1);
    float lse = mx + __logf(__expf(l0 - mx) + __expf(l1 - mx));
    s += lse - (label[e] ? l1 : l0);                    // NLL
  }
  s += __shfl_xor(s, 16);                               // both 16-lane halves
  if (lane == 0) atomicAdd(acc, s);
}

__global__ void gat_finalize(const float* __restrict__ acc, float* __restrict__ out) {
  if (threadIdx.x == 0 && blockIdx.x == 0) out[0] = acc[0] * (1.0f / (float)GAT_E);
}

// ---------------------------------------------------------------------------
extern "C" void kernel_launch(void* const* d_in, const int* in_sizes, int n_in,
                              void* d_out, int out_size, void* d_ws, size_t ws_size,
                              hipStream_t stream) {
  const float* x     = (const float*)d_in[0];
  const float* adj   = (const float*)d_in[1];
  const int*   L     = (const int*)d_in[2];
  const int*   R     = (const int*)d_in[3];
  const int*   label = (const int*)d_in[4];
  const float* W     = (const float*)d_in[5];
  const float* a1    = (const float*)d_in[6];
  const float* a2    = (const float*)d_in[7];
  const float* W1    = (const float*)d_in[8];
  const float* b1    = (const float*)d_in[9];
  const float* W2    = (const float*)d_in[10];
  const float* b2    = (const float*)d_in[11];
  float* out = (float*)d_out;

  // workspace carve-up (256B aligned)
  char* ws = (char*)d_ws;
  size_t off = 0;
  auto carve = [&](size_t bytes) { char* p = ws + off; off = (off + bytes + 255) & ~(size_t)255; return p; };
  _Float16* xh   = (_Float16*)carve((size_t)GAT_N * GAT_FEAT * 2);   // 3.0 MB
  _Float16* pW   = (_Float16*)carve(16 * 4 * 32 * 16 * 2);           // 64 KB
  _Float16* pW1  = (_Float16*)carve(4 * 4 * 32 * 16 * 2);            // 16 KB
  float*    Y    = (float*)carve((size_t)GAT_N * GAT_HD * 4);        // 768 KB
  float*    s1   = (float*)carve((size_t)GAT_H * GAT_N * 4);
  float*    s2   = (float*)carve((size_t)GAT_H * GAT_N * 4);
  _Float16* embh = (_Float16*)carve((size_t)GAT_N * GAT_HD * 2);
  float*    acc  = (float*)carve(256);
  (void)ws_size; (void)in_sizes; (void)n_in; (void)out_size;

  hipMemsetAsync(acc, 0, sizeof(float), stream);

  gat_cvt_x<<<(GAT_N * GAT_FEAT + 255) / 256, 256, 0, stream>>>(x, xh, GAT_N * GAT_FEAT);
  gat_pack_wcat<<<(16 * 4 * 32 * 16) / 256, 256, 0, stream>>>(W, pW);
  gat_pack_w1<<<(4 * 4 * 32 * 16) / 256, 256, 0, stream>>>(W1, pW1);

  gat_gemm_wh<<<192 / 8, 256, 0, stream>>>(xh, pW, Y);               // 192 waves
  gat_scores<<<(GAT_H * GAT_N + 255) / 256, 256, 0, stream>>>(Y, a1, a2, s1, s2);
  gat_attn<<<GAT_N, 256, 0, stream>>>(adj, Y, s1, s2, embh);

  int tiles = GAT_E / 16;                                            // 6250
  gat_classifier<<<(tiles + 7) / 8, 256, 0, stream>>>(embh, pW1, b1, W2, b2,
                                                      L, R, label, acc);
  gat_finalize<<<1, 1, 0, stream>>>(acc, out);
}
